// Generator_70978629534406
// MI455X (gfx1250) — matrix operand order repaired and checked by reading, not compiled, not run on hardware
//
#include <hip/hip_runtime.h>
#include <hip/hip_bf16.h>
#include <math.h>

typedef _Float16 v16h __attribute__((ext_vector_type(16)));
typedef _Float16 v8h  __attribute__((ext_vector_type(8)));
typedef float    v8f  __attribute__((ext_vector_type(8)));

#define B_SZ   8192
#define IN_SZ  128
#define H_SZ   256
#define OUT_SZ 2048
#define AR_N   7
#define ROWS   32
#define CHUNK  128
#define NCHUNK (OUT_SZ / CHUNK)   // 16

// ---------- helpers ----------

// XOR swizzle on 8-element (16B) chunks: conflict-free LDS row access
__device__ __forceinline__ int sw_idx(int row, int col, int ncols) {
    return row * ncols + ((((col >> 3) ^ (row & 15)) << 3) | (col & 7));
}

// skewed index for the AR buffer (float), conflict-free on both axes
__device__ __forceinline__ int ar_idx(int row, int t) {
    return row * CHUNK + ((t + 2 * row) & (CHUNK - 1));
}

__device__ __forceinline__ v8f zero8() {
    v8f z;
#pragma unroll
    for (int i = 0; i < 8; ++i) z[i] = 0.0f;
    return z;
}

// A fragment (16x32 f16): lane<16 row=l16, K = {0..7, 16..23}; lane>=16: {8..15, 24..31}
__device__ __forceinline__ v16h load_a_sw(const _Float16* lds, int ncols,
                                          int mBase, int kb, int hl, int l16) {
    int row = mBase + l16;
    int c0  = kb * 32 + hl * 8;
    v8h a0 = *(const v8h*)(lds + sw_idx(row, c0,      ncols));
    v8h a1 = *(const v8h*)(lds + sw_idx(row, c0 + 16, ncols));
    return __builtin_shufflevector(a0, a1, 0,1,2,3,4,5,6,7,8,9,10,11,12,13,14,15);
}

// B fragment (32x16 f16): lane<16 col=l16 K = {0..7, 16..23}; lane>=16: {8..15, 24..31}.
// Weights are row-major [N][K]: two 16B loads per lane.
__device__ __forceinline__ v16h load_b(const _Float16* __restrict__ w, int K,
                                       int nBase, int kb, int hl, int l16) {
    int col = nBase + l16;
    const _Float16* p = w + (size_t)col * K + kb * 32 + hl * 8;
    v8h b0 = *(const v8h*)(p);
    v8h b1 = *(const v8h*)(p + 16);
    return __builtin_shufflevector(b0, b1, 0,1,2,3,4,5,6,7,8,9,10,11,12,13,14,15);
}

// hi/lo split product: near-f32 accuracy with 3 f16 WMMAs
__device__ __forceinline__ v8f mma3(v16h ah, v16h al, v16h bh, v16h bl, v8f c) {
    c = __builtin_amdgcn_wmma_f32_16x16x32_f16(false, ah, false, bh, (short)0, c, false, false);
    c = __builtin_amdgcn_wmma_f32_16x16x32_f16(false, ah, false, bl, (short)0, c, false, false);
    c = __builtin_amdgcn_wmma_f32_16x16x32_f16(false, al, false, bh, (short)0, c, false, false);
    return c;
}

// one hidden layer: out = relu(A @ W^T + b), A in LDS (split), W in global (split)
__device__ __forceinline__ void gemm_layer(
    const _Float16* aHi, const _Float16* aLo, int aK,
    const _Float16* __restrict__ wHi, const _Float16* __restrict__ wLo,
    const float* __restrict__ bias,
    _Float16* oHi, _Float16* oLo,
    int wave, int hl, int l16)
{
    const int mt  = wave & 1;
    const int ntB = (wave >> 1) * 4;
    v8f acc[4];
#pragma unroll
    for (int i = 0; i < 4; ++i) acc[i] = zero8();
    for (int kb = 0; kb < aK / 32; ++kb) {
        v16h ah = load_a_sw(aHi, aK, mt * 16, kb, hl, l16);
        v16h al = load_a_sw(aLo, aK, mt * 16, kb, hl, l16);
#pragma unroll
        for (int i = 0; i < 4; ++i) {
            v16h bh = load_b(wHi, aK, (ntB + i) * 16, kb, hl, l16);
            v16h bl = load_b(wLo, aK, (ntB + i) * 16, kb, hl, l16);
            acc[i] = mma3(ah, al, bh, bl, acc[i]);
        }
    }
#pragma unroll
    for (int i = 0; i < 4; ++i) {
        const int n = (ntB + i) * 16 + l16;
        float bv = bias[n];
#pragma unroll
        for (int r = 0; r < 8; ++r) {
            int row = mt * 16 + hl * 8 + r;   // C/D layout: lane<16 M=r, lane>=16 M=r+8
            float v = fmaxf(acc[i][r] + bv, 0.0f);
            _Float16 hi = (_Float16)v;
            oHi[sw_idx(row, n, H_SZ)] = hi;
            oLo[sw_idx(row, n, H_SZ)] = (_Float16)(v - (float)hi);
        }
    }
}

// one AR(7) chunk for one row (lane): single dependent FMA on the critical path
__device__ __forceinline__ void ar_chunk(const float* __restrict__ np,
                                         const float* kc, float* cr,
                                         float* arb, int r, int c)
{
    const int tBase = c * CHUNK;
#pragma unroll 4
    for (int u = 0; u < CHUNK; ++u) {
        int t = tBase + u;
        float part = np[t] + cr[1]*kc[1] + cr[2]*kc[2] + cr[3]*kc[3]
                   + cr[4]*kc[4] + cr[5]*kc[5] + cr[6]*kc[6];
        float val = fmaf(cr[0], kc[0], part);
        if (t < AR_N) val = 0.0f;
        arb[ar_idx(r, u)] = val;
        cr[6]=cr[5]; cr[5]=cr[4]; cr[4]=cr[3]; cr[3]=cr[2]; cr[2]=cr[1]; cr[1]=cr[0];
        cr[0]=val;
    }
}

// ---------- kernels ----------

// split f32 weights into f16 hi + f16 lo in workspace
__global__ __launch_bounds__(256) void prep_split(
    const float* __restrict__ W1, const float* __restrict__ W2,
    const float* __restrict__ W3, _Float16* __restrict__ ws)
{
    const int N1 = H_SZ * IN_SZ, N2 = H_SZ * H_SZ, N3 = OUT_SZ * H_SZ;
    int i = blockIdx.x * 256 + threadIdx.x;
    const float* src; _Float16* hi; _Float16* lo; int j;
    if (i < N1)                 { src = W1; j = i;           hi = ws;               lo = hi + N1; }
    else if (i < N1 + N2)       { src = W2; j = i - N1;      hi = ws + 2*N1;        lo = hi + N2; }
    else if (i < N1 + N2 + N3)  { src = W3; j = i - N1 - N2; hi = ws + 2*N1 + 2*N2; lo = hi + N3; }
    else return;
    float v = src[j];
    _Float16 h = (_Float16)v;
    hi[j] = h;
    lo[j] = (_Float16)(v - (float)h);
}

__global__ __launch_bounds__(256) void mlp_fused(
    const float* __restrict__ x,
    const float* __restrict__ noise,
    const float* __restrict__ b1,
    const float* __restrict__ b2,
    const float* __restrict__ b3,
    const float* __restrict__ arc,
    const _Float16* __restrict__ w1hi, const _Float16* __restrict__ w1lo,
    const _Float16* __restrict__ w2hi, const _Float16* __restrict__ w2lo,
    const _Float16* __restrict__ w3hi, const _Float16* __restrict__ w3lo,
    float* __restrict__ out)
{
    // 64 KiB static LDS, two 32 KiB regions with phase overlays
    __shared__ __align__(32) unsigned char region1[ROWS * H_SZ * 2 * 2];
    __shared__ __align__(32) unsigned char region2[ROWS * H_SZ * 2 * 2];

    _Float16* h1hi = (_Float16*)region1;           // phase 1-2
    _Float16* h1lo = h1hi + ROWS * H_SZ;
    float*    arb  = (float*)region1;              // phase 3: [2][ROWS][CHUNK]

    _Float16* xhi  = (_Float16*)region2;           // phase 0-1
    _Float16* xlo  = xhi + ROWS * IN_SZ;
    _Float16* h2hi = (_Float16*)region2;           // phase 2-3 (x is dead)
    _Float16* h2lo = h2hi + ROWS * H_SZ;

    const int tid  = threadIdx.x;
    const int wave = tid >> 5;
    const int lane = tid & 31;
    const int hl   = lane >> 4;
    const int l16  = lane & 15;
    const int rowBase = blockIdx.x * ROWS;

    // stage x tile, split to f16 hi/lo (coalesced global read)
    for (int i = tid; i < ROWS * IN_SZ; i += 256) {
        int r = i >> 7, cl = i & (IN_SZ - 1);
        float v = x[(size_t)(rowBase + r) * IN_SZ + cl];
        _Float16 hi = (_Float16)v;
        xhi[sw_idx(r, cl, IN_SZ)] = hi;
        xlo[sw_idx(r, cl, IN_SZ)] = (_Float16)(v - (float)hi);
    }
    __syncthreads();

    gemm_layer(xhi,  xlo,  IN_SZ, w1hi, w1lo, b1, h1hi, h1lo, wave, hl, l16);
    __syncthreads();
    gemm_layer(h1hi, h1lo, H_SZ,  w2hi, w2lo, b2, h2hi, h2lo, wave, hl, l16);
    __syncthreads();

    // ---- GEMM3 (waves 0-6) pipelined with AR(7) noise (wave 7) ----
    float cr[AR_N], kc[AR_N];
#pragma unroll
    for (int i = 0; i < AR_N; ++i) { cr[i] = 0.0f; kc[i] = arc[i]; }
    const float* np = noise + (size_t)(rowBase + lane) * OUT_SZ;

    if (wave == 7) ar_chunk(np, kc, cr, arb, lane, 0);
    __syncthreads();

    for (int c = 0; c < NCHUNK; ++c) {
        float* arRead  = arb + (c & 1) * (ROWS * CHUNK);
        float* arWrite = arb + ((c + 1) & 1) * (ROWS * CHUNK);
        if (wave == 7) {
            if (c + 1 < NCHUNK) ar_chunk(np, kc, cr, arWrite, lane, c + 1);
        } else {
            for (int tIdx = wave; tIdx < 16; tIdx += 7) {
                const int mt = tIdx & 1;
                const int n0 = (tIdx >> 1) * 16;          // col within chunk
                v8f acc = zero8();
                for (int kb = 0; kb < H_SZ / 32; ++kb) {
                    v16h ah = load_a_sw(h2hi, H_SZ, mt * 16, kb, hl, l16);
                    v16h al = load_a_sw(h2lo, H_SZ, mt * 16, kb, hl, l16);
                    v16h bh = load_b(w3hi, H_SZ, c * CHUNK + n0, kb, hl, l16);
                    v16h bl = load_b(w3lo, H_SZ, c * CHUNK + n0, kb, hl, l16);
                    acc = mma3(ah, al, bh, bl, acc);
                }
                const int nIn = n0 + l16;
                const int nGl = c * CHUNK + nIn;
                float bv = b3[nGl];
#pragma unroll
                for (int r = 0; r < 8; ++r) {
                    int rowL = mt * 16 + hl * 8 + r;
                    float v = tanhf(acc[r] + bv) + arRead[ar_idx(rowL, nIn)];
                    out[(size_t)(rowBase + rowL) * OUT_SZ + nGl] = v;
                }
            }
        }
        __syncthreads();
    }
}

// ---------- launcher ----------

extern "C" void kernel_launch(void* const* d_in, const int* in_sizes, int n_in,
                              void* d_out, int out_size, void* d_ws, size_t ws_size,
                              hipStream_t stream) {
    const float* x   = (const float*)d_in[0];
    const float* nz  = (const float*)d_in[1];
    const float* W1  = (const float*)d_in[2];
    const float* b1  = (const float*)d_in[3];
    const float* W2  = (const float*)d_in[4];
    const float* b2  = (const float*)d_in[5];
    const float* W3  = (const float*)d_in[6];
    const float* b3  = (const float*)d_in[7];
    const float* arc = (const float*)d_in[8];
    float* out = (float*)d_out;
    _Float16* ws = (_Float16*)d_ws;

    const int N1 = H_SZ * IN_SZ, N2 = H_SZ * H_SZ, N3 = OUT_SZ * H_SZ;
    _Float16* w1hi = ws;
    _Float16* w1lo = w1hi + N1;
    _Float16* w2hi = w1lo + N1;
    _Float16* w2lo = w2hi + N2;
    _Float16* w3hi = w2lo + N2;
    _Float16* w3lo = w3hi + N3;

    const int tot = N1 + N2 + N3;   // 622592, multiple of 256
    prep_split<<<(tot + 255) / 256, 256, 0, stream>>>(W1, W2, W3, ws);
    mlp_fused<<<B_SZ / ROWS, 256, 0, stream>>>(x, nz, b1, b2, b3, arc,
        w1hi, w1lo, w2hi, w2lo, w3hi, w3lo, out);
}
